// GMML_64166811402428
// MI455X (gfx1250) — matrix operand order repaired and checked by reading, not compiled, not run on hardware
//
#include <hip/hip_runtime.h>
#include <math.h>
#include <stdint.h>

// Problem constants (from reference): N=2048 points, S=64, G=1, D=512.
#define DD    512
#define NPTS  2048
#define NS    64
#define MAT   (DD*DD)           // 262144 floats per matrix

typedef __attribute__((ext_vector_type(2)))  float  v2f;
typedef __attribute__((ext_vector_type(8)))  float  v8f;
typedef __attribute__((ext_vector_type(16))) __bf16 v16bf;

static __device__ __forceinline__ v8f wmma4(v2f a, v2f b, v8f c) {
  // D = A(16x4,f32) * B(4x16,f32) + C(16x16,f32)
  return __builtin_amdgcn_wmma_f32_16x16x4_f32(
      /*neg_a=*/false, a, /*neg_b=*/false, b,
      /*c_mod=*/(short)0, c, /*reuse_a=*/false, /*reuse_b=*/false);
}

static __device__ __forceinline__ v8f wmma_bf16(v16bf a, v16bf b, v8f c) {
  // D = A(16x32,bf16) * B(32x16,bf16) + C(16x16,f32) -- the fast XDL pipe
  return __builtin_amdgcn_wmma_f32_16x16x32_bf16(
      /*neg_a=*/false, a, /*neg_b=*/false, b,
      /*c_mod=*/(short)0, c, /*reuse_a=*/false, /*reuse_b=*/false);
}

// Async global->LDS B128 copy (CDNA5 TDM-lite path, ASYNCcnt-tracked).
// ldsoff = byte offset in workgroup LDS (flat addr truncated per ISA 10.2).
static __device__ __forceinline__ void async_g2l_b128(uint32_t ldsoff, const void* gptr) {
  asm volatile("global_load_async_to_lds_b128 %0, %1, off"
               :: "v"(ldsoff), "v"((uint64_t)(uintptr_t)gptr)
               : "memory");
}
static __device__ __forceinline__ void wait_asynccnt0() {
  asm volatile("s_wait_asynccnt 0" ::: "memory");
}

// ---------------------------------------------------------------------------
// Kernel 1: m[s] = P[s]^T P[s]   (P = sigma_p[s,0], row-major 512x512)
// f32 WMMA (precision feeds the eigensolver / Cholesky).
// Tile 32(i) x 64(j) per block; 8 waves, each owns one 16x16 subtile.
// B-tile is staged via async global->LDS B128 DMA; A-tile via normal ds path.
// ---------------------------------------------------------------------------
#define BS_PITCH 68   // floats per Bs row: 272B, 16B-aligned for async B128
__global__ void ata_kernel(const float* __restrict__ P, float* __restrict__ M) {
  const int s    = blockIdx.z;
  const int i0   = blockIdx.x * 32;
  const int j0   = blockIdx.y * 64;
  const int tid  = threadIdx.x;
  const int lane = tid & 31;
  const int wave = tid >> 5;
  const float* Ps = P + (size_t)s * MAT;
  float*       Ms = M + (size_t)s * MAT;

  __shared__ __align__(16) float As[64 * 33];        // [k][i-local], stride 33
  __shared__ __align__(16) float Bs[64 * BS_PITCH];  // [k][j-local], stride 68

  const int i_sub = (wave >> 2) * 16;  // 0 or 16
  const int j_sub = (wave & 3) * 16;   // 0,16,32,48
  const int half  = lane >> 4;         // k-pair selector per ISA A/B layout
  const int ml    = lane & 15;

  v8f acc = {};
  for (int k0 = 0; k0 < DD; k0 += 64) {
    __syncthreads();
    // ---- async DMA: Bs[k][c] <- P[k0+k][j0+c], 16B per lane per issue ----
    for (int l = tid; l < 64 * 16; l += 256) {
      const int k = l >> 4;
      const int c = (l & 15) * 4;
      const uint32_t ldsoff = (uint32_t)(uintptr_t)&Bs[k * BS_PITCH + c];
      async_g2l_b128(ldsoff, Ps + (size_t)(k0 + k) * DD + j0 + c);
    }
    // ---- regular path for the A tile ----
    for (int l = tid; l < 64 * 32; l += 256) {
      int k = l >> 5, c = l & 31;
      As[k * 33 + c] = Ps[(size_t)(k0 + k) * DD + i0 + c];
    }
    if (k0 + 64 < DD)  // pull next K-chunk toward L2/WGP$ (global_prefetch_b8)
      __builtin_prefetch(Ps + (size_t)(k0 + 64 + (tid >> 2)) * DD + i0, 0, 1);
    wait_asynccnt0();  // our async stores to LDS are visible; barrier publishes all
    __syncthreads();
#pragma unroll 4
    for (int kk = 0; kk < 64; kk += 4) {
      const int kr = kk + 2 * half;
      v2f a, b;
      a.x = As[kr * 33 + i_sub + ml];              // A[m][k] = P[k][i]
      a.y = As[(kr + 1) * 33 + i_sub + ml];
      b.x = Bs[kr * BS_PITCH + j_sub + ml];        // B[k][n] = P[k][j]
      b.y = Bs[(kr + 1) * BS_PITCH + j_sub + ml];
      acc = wmma4(a, b, acc);
    }
  }
#pragma unroll
  for (int r = 0; r < 8; ++r) {
    const int ii = i0 + i_sub + r + 8 * half;  // C/D layout: VGPR r -> M=r / M=8+r
    const int jj = j0 + j_sub + ml;
    Ms[(size_t)ii * DD + jj] = acc[r];
  }
}

// ---------------------------------------------------------------------------
// Kernel 2: Householder tridiagonalization of m[s] (copy into T), one WG/s.
// ---------------------------------------------------------------------------
__global__ void tridiag_kernel(const float* __restrict__ Mbuf, float* __restrict__ Tbuf,
                               float* __restrict__ d_arr, float* __restrict__ e_arr) {
  const int s   = blockIdx.x;
  const int tid = threadIdx.x;  // 256
  const float* Msrc = Mbuf + (size_t)s * MAT;
  float*       T    = Tbuf + (size_t)s * MAT;

  __shared__ float v[DD];
  __shared__ float q[DD];
  __shared__ float red[256];

  for (int l = tid; l < MAT; l += 256) T[l] = Msrc[l];
  __syncthreads();

  for (int k = 0; k <= DD - 3; ++k) {
    float ss = 0.f;
    for (int i = k + 1 + tid; i < DD; i += 256) { float t = T[i * DD + k]; ss += t * t; }
    red[tid] = ss; __syncthreads();
    for (int off = 128; off > 0; off >>= 1) { if (tid < off) red[tid] += red[tid + off]; __syncthreads(); }
    const float sigma = red[0];
    const float t0    = T[(k + 1) * DD + k];
    const float nrm   = sqrtf(sigma);
    const float alpha = (t0 >= 0.f) ? -nrm : nrm;
    const bool  skip  = !(sigma > 1e-24f);
    const float uinv  = skip ? 0.f : rsqrtf(2.0f * (sigma - alpha * t0));
    __syncthreads();

    for (int i = tid; i < DD; i += 256) {
      float ui = 0.f;
      if (i > k) ui = (T[i * DD + k] - (i == k + 1 ? alpha : 0.f)) * uinv;
      v[i] = ui;
    }
    __syncthreads();

    if (!skip) {
      for (int i = k + 1 + tid; i < DD; i += 256) {
        const float* row = T + (size_t)i * DD;
        float acc = 0.f;
        for (int j = k + 1; j < DD; ++j) acc += row[j] * v[j];
        q[i] = acc;
      }
      __syncthreads();
      float lp = 0.f;
      for (int i = k + 1 + tid; i < DD; i += 256) lp += v[i] * q[i];
      red[tid] = lp; __syncthreads();
      for (int off = 128; off > 0; off >>= 1) { if (tid < off) red[tid] += red[tid + off]; __syncthreads(); }
      const float vp = red[0];
      __syncthreads();
      for (int i = k + 1 + tid; i < DD; i += 256) q[i] -= vp * v[i];
      __syncthreads();
      for (int i = k + 1 + tid; i < DD; i += 256) {
        const float vi2 = 2.f * v[i], qi2 = 2.f * q[i];
        float* row = T + (size_t)i * DD;
        for (int j = k + 1; j < DD; ++j) row[j] -= vi2 * q[j] + qi2 * v[j];
      }
    }
    if (tid == 0) e_arr[s * DD + k] = skip ? t0 : alpha;
    __syncthreads();
  }
  for (int i = tid; i < DD; i += 256) d_arr[s * DD + i] = T[(size_t)i * DD + i];
  if (tid == 0) e_arr[s * DD + (DD - 2)] = T[(size_t)(DD - 1) * DD + (DD - 2)];
}

// ---------------------------------------------------------------------------
// Kernel 3: Sturm-sequence bisection; thread t finds the t-th eigenvalue.
// ---------------------------------------------------------------------------
__device__ __forceinline__ int sturm_count(const float* dsh, const float* e2, float x) {
  int cnt = 0;
  float qq = dsh[0] - x;
  if (qq < 0.f) cnt++;
  for (int i = 1; i < DD; ++i) {
    float den = qq;
    if (fabsf(den) < 1e-30f) den = (den < 0.f) ? -1e-30f : 1e-30f;
    qq = dsh[i] - x - e2[i - 1] / den;
    if (qq < 0.f) cnt++;
  }
  return cnt;
}

__global__ void eig_bisect_kernel(const float* __restrict__ d_arr,
                                  const float* __restrict__ e_arr,
                                  float* __restrict__ eig) {
  const int s = blockIdx.x;
  const int t = threadIdx.x;  // 512
  __shared__ float dsh[DD];
  __shared__ float e2[DD];
  dsh[t] = d_arr[s * DD + t];
  const float ev = (t < DD - 1) ? e_arr[s * DD + t] : 0.f;
  e2[t] = ev * ev;
  __syncthreads();

  float lo = 1e30f, hi = -1e30f;
  for (int i = 0; i < DD; ++i) {
    const float r = ((i > 0) ? sqrtf(e2[i - 1]) : 0.f) + sqrtf(e2[i]);
    lo = fminf(lo, dsh[i] - r);
    hi = fmaxf(hi, dsh[i] + r);
  }
  for (int it = 0; it < 60; ++it) {
    const float mid = 0.5f * (lo + hi);
    if (sturm_count(dsh, e2, mid) <= t) lo = mid; else hi = mid;
  }
  eig[s * DD + t] = 0.5f * (lo + hi);
}

// ---------------------------------------------------------------------------
// Kernel 4: sigma = m + 0.01*eig_i on diag; in-place Cholesky (lower); logdet.
// ---------------------------------------------------------------------------
__global__ void chol_kernel(float* __restrict__ Mbuf, const float* __restrict__ eig,
                            float* __restrict__ ldet) {
  const int s   = blockIdx.x;
  const int tid = threadIdx.x;  // 256
  float* A = Mbuf + (size_t)s * MAT;
  __shared__ float colk[DD];
  __shared__ float red[256];
  __shared__ float dinv_sh;

  for (int i = tid; i < DD; i += 256) A[(size_t)i * DD + i] += 0.01f * eig[s * DD + i];
  __syncthreads();

  for (int k = 0; k < DD; ++k) {
    if (tid == 0) {
      const float dv = sqrtf(A[(size_t)k * DD + k]);
      A[(size_t)k * DD + k] = dv;
      dinv_sh = 1.0f / dv;
      colk[k] = dv;
    }
    __syncthreads();
    const float dinv = dinv_sh;
    for (int i = k + 1 + tid; i < DD; i += 256) {
      const float val = A[(size_t)i * DD + k] * dinv;
      A[(size_t)i * DD + k] = val;
      colk[i] = val;
    }
    __syncthreads();
    for (int i = k + 1 + tid; i < DD; i += 256) {
      const float ci = colk[i];
      float* row = A + (size_t)i * DD;
      for (int j = k + 1; j <= i; ++j) row[j] -= ci * colk[j];
    }
    __syncthreads();
  }
  float ls = 0.f;
  for (int i = tid; i < DD; i += 256) ls += logf(A[(size_t)i * DD + i]);
  red[tid] = ls; __syncthreads();
  for (int off = 128; off > 0; off >>= 1) { if (tid < off) red[tid] += red[tid + off]; __syncthreads(); }
  if (tid == 0) ldet[s] = 2.0f * red[0];
}

// ---------------------------------------------------------------------------
// Kernel 5: Linv = L^{-1} (lower-triangular inverse), thread j owns column j.
// ---------------------------------------------------------------------------
__global__ void trinv_kernel(const float* __restrict__ Mbuf, float* __restrict__ Linv) {
  const int s = blockIdx.x;
  const int j = threadIdx.x;  // 512
  const float* L = Mbuf + (size_t)s * MAT;
  float*       V = Linv + (size_t)s * MAT;
  for (int i = 0; i < j; ++i) V[(size_t)i * DD + j] = 0.f;
  V[(size_t)j * DD + j] = 1.0f / L[(size_t)j * DD + j];
  for (int i = j + 1; i < DD; ++i) {
    const float* row = L + (size_t)i * DD;
    float acc = 0.f;
    for (int k = j; k < i; ++k) acc += row[k] * V[(size_t)k * DD + j];
    V[(size_t)i * DD + j] = -acc / row[i];
  }
}

// ---------------------------------------------------------------------------
// Kernel 6: Z = (x - mu[s]) * Linv^T via bf16x3 split WMMA (hi*hi+hi*lo+lo*hi),
// near-f32 precision on the 8x-throughput 16x16x32 bf16 pipe.
// maha = rowsum(Z^2); fused logp epilogue, transposed store.
// 303KB dynamic LDS: whole 128-row x-block (hi+lo) stays resident.
// Grid (NPTS/128, S); 8 waves, wave w owns rows [w*16, w*16+16).
// ---------------------------------------------------------------------------
#define XH_STRIDE 520   // bf16 elements per Xhi/Xlo row
#define BH_STRIDE 18    // bf16 elements per Bhi/Blo row
#define MAHA_LDS_BYTES ((2 * 128 * XH_STRIDE + 2 * DD * BH_STRIDE) * sizeof(__bf16))

__global__ void maha_kernel(const float* __restrict__ x, const float* __restrict__ mu,
                            const float* __restrict__ Linv, const float* __restrict__ ldet,
                            float* __restrict__ out) {
  const int s    = blockIdx.y;
  const int n0   = blockIdx.x * 128;
  const int tid  = threadIdx.x;
  const int lane = tid & 31;
  const int wave = tid >> 5;

  extern __shared__ __bf16 smem_bf[];
  __bf16* Xhi = smem_bf;                    // [128][520]
  __bf16* Xlo = Xhi + 128 * XH_STRIDE;      // [128][520]
  __bf16* Bhi = Xlo + 128 * XH_STRIDE;      // [512][18]
  __bf16* Blo = Bhi + DD * BH_STRIDE;       // [512][18]

  const float* LT = Linv + (size_t)s * MAT;

  // center x and split into bf16 hi/lo once per block
  for (int l = tid; l < 128 * DD; l += 256) {
    const int r = l >> 9, c = l & (DD - 1);
    const float xc = x[(size_t)(n0 + r) * DD + c] - mu[s * DD + c];
    const __bf16 h = (__bf16)xc;
    Xhi[r * XH_STRIDE + c] = h;
    Xlo[r * XH_STRIDE + c] = (__bf16)(xc - (float)h);
  }

  const int msub = wave * 16;
  const int half = lane >> 4;
  const int ml   = lane & 15;
  float ssum[8] = {0.f, 0.f, 0.f, 0.f, 0.f, 0.f, 0.f, 0.f};

  for (int j0 = 0; j0 < DD; j0 += 16) {
    __syncthreads();  // covers initial X fill and prior tile's compute
    {
      // stage B[k][n] = Linv[j0+n][k], split hi/lo
      const int n  = tid >> 4;
      const int kb = (tid & 15) * 32;
      const float* lr = LT + (size_t)(j0 + n) * DD + kb;
#pragma unroll 8
      for (int k = 0; k < 32; ++k) {
        const float v = lr[k];
        const __bf16 h = (__bf16)v;
        Bhi[(kb + k) * BH_STRIDE + n] = h;
        Blo[(kb + k) * BH_STRIDE + n] = (__bf16)(v - (float)h);
      }
      if (j0 + 16 < DD)  // prefetch next j-tile of Linv (global_prefetch_b8)
        __builtin_prefetch(LT + (size_t)(j0 + 16 + n) * DD + kb, 0, 1);
    }
    __syncthreads();

    v8f acc = {};
    const __bf16* XhR = Xhi + (msub + ml) * XH_STRIDE;
    const __bf16* XlR = Xlo + (msub + ml) * XH_STRIDE;
    for (int k0 = 0; k0 < DD; k0 += 32) {
      v16bf ah, al, bh, bl;
#pragma unroll
      for (int e = 0; e < 16; ++e) {
        // A 16x32 bf16 layout: VGPR j holds K=2j..2j+1 (half0: K 0-7/16-23;
        // half1: K 8-15/24-31). B 32x16: same K order as A.
        const int ka = k0 + e + ((e < 8) ? 8 * half : 8 + 8 * half);
        ah[e] = XhR[ka];
        al[e] = XlR[ka];
        const int kb2 = k0 + e + ((e < 8) ? 8 * half : 8 + 8 * half);
        bh[e] = Bhi[kb2 * BH_STRIDE + ml];
        bl[e] = Blo[kb2 * BH_STRIDE + ml];
      }
      acc = wmma_bf16(ah, bh, acc);   // hi*hi
      acc = wmma_bf16(ah, bl, acc);   // hi*lo
      acc = wmma_bf16(al, bh, acc);   // lo*hi   (lo*lo ~2^-18, dropped)
    }
#pragma unroll
    for (int r = 0; r < 8; ++r) ssum[r] += acc[r] * acc[r];
  }

  const float cst = 940.9930579935848f;  // 512 * log(2*pi)
  const float ld  = ldet[s];
#pragma unroll
  for (int r = 0; r < 8; ++r) {
    float t = ssum[r];
    t += __shfl_xor(t, 1, 32);
    t += __shfl_xor(t, 2, 32);
    t += __shfl_xor(t, 4, 32);
    t += __shfl_xor(t, 8, 32);      // sum over the 16 N-lanes of this half
    if (ml == 0) {
      const int Mrow = msub + r + 8 * half;
      out[(size_t)(n0 + Mrow) * NS + s] = -0.5f * (t + ld + cst);
    }
  }
}

// ---------------------------------------------------------------------------
// Host-side launch. Inputs: [0]=x (N,D) f32, [1]=mu (S,1,D) f32,
// [2]=omega (unused), [3]=sigma_p (S,1,D,D) f32. Output: logp.T (N,S) f32.
// Workspace: M (64MB) | T/Linv (64MB) | d | e | eig | logdet.
// ---------------------------------------------------------------------------
extern "C" void kernel_launch(void* const* d_in, const int* in_sizes, int n_in,
                              void* d_out, int out_size, void* d_ws, size_t ws_size,
                              hipStream_t stream) {
  (void)in_sizes; (void)n_in; (void)out_size; (void)ws_size;
  const float* x       = (const float*)d_in[0];
  const float* mu      = (const float*)d_in[1];
  const float* sigma_p = (const float*)d_in[3];
  float* out = (float*)d_out;

  float* W    = (float*)d_ws;
  float* Mbuf = W;                                   // 64 * 512*512
  float* Tbuf = W + (size_t)NS * MAT;                // tridiag scratch, then Linv
  float* Darr = W + (size_t)2 * NS * MAT;
  float* Earr = Darr + NS * DD;
  float* Eig  = Earr + NS * DD;
  float* Ldet = Eig + NS * DD;

  // 1) m = P^T P (batched f32 WMMA GEMM, async-LDS staged B tiles)
  ata_kernel<<<dim3(DD / 32, DD / 64, NS), 256, 0, stream>>>(sigma_p, Mbuf);
  // 2) tridiagonalize each m
  tridiag_kernel<<<NS, 256, 0, stream>>>(Mbuf, Tbuf, Darr, Earr);
  // 3) all 512 eigenvalues per matrix via Sturm bisection
  eig_bisect_kernel<<<NS, DD, 0, stream>>>(Darr, Earr, Eig);
  // 4) sigma = m + 0.01*diag(eig); Cholesky in place; logdet
  chol_kernel<<<NS, 256, 0, stream>>>(Mbuf, Eig, Ldet);
  // 5) Linv = L^{-1} (overwrites Tbuf)
  trinv_kernel<<<NS, DD, 0, stream>>>(Mbuf, Tbuf);
  // 6) maha GEMM on bf16x3 split WMMA + fused logp epilogue
  maha_kernel<<<dim3(NPTS / 128, NS), 256, MAHA_LDS_BYTES, stream>>>(x, mu, Tbuf, Ldet, out);
}
